// NeuralODE_46823733461180
// MI455X (gfx1250) — matrix operand order repaired and checked by reading, not compiled, not run on hardware
//
#include <hip/hip_runtime.h>
#include <hip/hip_bf16.h>
#include <stdint.h>

// ---------------- types ----------------
typedef __attribute__((ext_vector_type(16))) _Float16 v16h;
typedef __attribute__((ext_vector_type(8)))  _Float16 v8h;
typedef __attribute__((ext_vector_type(8)))  float    v8f;
typedef __attribute__((ext_vector_type(4)))  float    v4f;
typedef __attribute__((ext_vector_type(4)))  unsigned int u32x4;
typedef __attribute__((ext_vector_type(8)))  int      i32x8;
typedef __attribute__((ext_vector_type(4)))  int      i32x4;

#define TT   32
#define BB   4096
#define DD   32
#define WWID 64
#define KSUB 2

// LDS layout (byte offsets inside the single static __shared__ block; block is
// the only LDS allocation so its base offset is 0, which the TDM descriptors rely on)
#define ACTA_OFF   0        // 16x32 f16, y-stage in B-operand layout, col-major [n][k]
#define ACTB_OFF   1024     // 16x64 f16, hidden activations, col-major [n][k]
#define BIAS_OFF   3072     // 160 f32: b1[64] | b2[64] | b3[32]
#define KBUF_OFF   3712     // 6 * 512 f32: Dopri5 stage derivatives k1..k6
#define SMEM_BYTES 16000

// ---------------- small helpers ----------------
__device__ __forceinline__ v8f wmma16(v16h a, v16h b, v8f c) {
  // v_wmma_f32_16x16x32_f16: D = A(16x32,f16) x B(32x16,f16) + C(16x16,f32)
  return __builtin_amdgcn_wmma_f32_16x16x32_f16(false, a, false, b, (short)0, c,
                                                false, false);
}

__device__ __forceinline__ float softplus_f(float x) {
  // stable softplus: max(x,0) + log1p(exp(-|x|)).
  // Raw v_exp_f32 / v_log_f32 (base-2) without OCML denorm fixups:
  //   exp2 arg <= 0 (underflow->0 is exactly right), log arg in (1,2] (never denorm).
  float e = __builtin_amdgcn_exp2f(-fabsf(x) * 1.4426950408889634f);
  float l = __builtin_amdgcn_logf(1.0f + e) * 0.6931471805599453f;
  return fmaxf(x, 0.0f) + l;
}

// Load one 16x32 f16 A-operand tile of a row-major weight matrix.
// A layout: lane m<16 = row m; halfs[0..15] = K k0..k0+15
// with k0 = kchunk*32 + (lane>=16)*16.
__device__ __forceinline__ v16h load_wtile(const float* __restrict__ Wm, int ncols,
                                           int mt, int kc, int n, int hi) {
  const float* p = Wm + (mt * 16 + n) * ncols + kc * 32 + hi * 16;
  v4f p0 = *(const v4f*)(p);
  v4f p1 = *(const v4f*)(p + 4);
  v4f p2 = *(const v4f*)(p + 8);
  v4f p3 = *(const v4f*)(p + 12);
  v16h a;
#pragma unroll
  for (int j = 0; j < 4; ++j) {
    a[j]      = (_Float16)p0[j];
    a[4 + j]  = (_Float16)p1[j];
    a[8 + j]  = (_Float16)p2[j];
    a[12 + j] = (_Float16)p3[j];
  }
  return a;
}

// B operand: col-major LDS activation buffer act[n*K + k]; one contiguous 32B read.
__device__ __forceinline__ v16h load_b_op(const _Float16* act, int K, int chunk,
                                          int n, int hi) {
  return *(const v16h*)(act + n * K + chunk * 32 + hi * 16);
}

// Store a C/D tile (8 f32 per lane, feature rows t*16+hi*8 .. +7, batch col n)
// as f16 into the col-major activation buffer: one contiguous 16B ds_store_b128.
__device__ __forceinline__ void store_act_tile(_Float16* act, int K, int t, int n,
                                               int hi, v8f v) {
  v8h ph;
#pragma unroll
  for (int r = 0; r < 8; ++r) ph[r] = (_Float16)v[r];
  *(v8h*)(act + n * K + t * 16 + hi * 8) = ph;
}

__device__ __forceinline__ v8f load_bias_tile(const float* bias, int off, int t,
                                              int hi) {
  const float* p = bias + off + t * 16 + hi * 8;
  v4f a = *(const v4f*)p;
  v4f b = *(const v4f*)(p + 4);
  v8f r;
#pragma unroll
  for (int j = 0; j < 4; ++j) { r[j] = a[j]; r[4 + j] = b[j]; }
  return r;
}

__device__ __forceinline__ void kstore(float* kbuf, int s, int t, int n, int hi,
                                       v8f v) {
  float* p = kbuf + (((s * 2 + t) * 2 + hi) * 16 + n) * 8;
  v4f a, b;
#pragma unroll
  for (int j = 0; j < 4; ++j) { a[j] = v[j]; b[j] = v[4 + j]; }
  *(v4f*)p = a;
  *(v4f*)(p + 4) = b;
}

__device__ __forceinline__ v8f kload(const float* kbuf, int s, int t, int n, int hi) {
  const float* p = kbuf + (((s * 2 + t) * 2 + hi) * 16 + n) * 8;
  v4f a = *(const v4f*)p;
  v4f b = *(const v4f*)(p + 4);
  v8f r;
#pragma unroll
  for (int j = 0; j < 4; ++j) { r[j] = a[j]; r[4 + j] = b[j]; }
  return r;
}

// ---------------- TDM: DMA a 1-D f32 vector into LDS ----------------
#if __has_builtin(__builtin_amdgcn_tensor_load_to_lds)
#define HAVE_TDM 1
__device__ __forceinline__ void tdm_load_1d(const void* src, unsigned lds_off,
                                            int nelem) {
  unsigned long long ga = (unsigned long long)(uintptr_t)src;
  u32x4 g0;
  i32x8 g1;
  i32x4 gz4;
  i32x8 gz8;
  gz4[0] = 0; gz4[1] = 0; gz4[2] = 0; gz4[3] = 0;
#pragma unroll
  for (int j = 0; j < 8; ++j) gz8[j] = 0;
  g0[0] = 1u;                                          // count=1, user descriptor
  g0[1] = lds_off;                                     // lds_addr
  g0[2] = (unsigned)(ga & 0xffffffffull);              // global_addr[31:0]
  g0[3] = (unsigned)((ga >> 32) & 0x01ffffffull) | (2u << 30);  // [56:32] | type=2
  g1[0] = (int)(2u << 16);                             // data_size = 4 bytes
  g1[1] = (int)((unsigned)nelem << 16);                // tensor_dim0[15:0] @ bits63:48
  g1[2] = (int)(1u << 16);                             // tensor_dim1 = 1 @ bits95:80
  g1[3] = (int)((unsigned)nelem << 16);                // tile_dim0 @ bits127:112
  g1[4] = 1;                                           // tile_dim1 = 1
  g1[5] = nelem;                                       // tensor_dim0_stride[31:0]
  g1[6] = (int)((unsigned)nelem << 16);                // tensor_dim1_stride[15:0]
  g1[7] = 0;
  // amdgpu-toolchain (clang-23) 6-arg form:
  // (g0:u32x4, g1:i32x8, g2:i32x4, g3:i32x4, extra:i32x8, cpol:i32)
  __builtin_amdgcn_tensor_load_to_lds(g0, g1, gz4, gz4, gz8, 0);
}
#else
#define HAVE_TDM 0
#endif

// ---------------- one MLP evaluation: k_s = f(y_stage in actA) ----------------
__device__ __forceinline__ void feval(const v16h A1[4], const v16h A2[4][2],
                                      const v16h A3[2][2], _Float16* actA,
                                      _Float16* actB, const float* bias,
                                      float* kbuf, int s, int n, int hi) {
  v8f zero = {0, 0, 0, 0, 0, 0, 0, 0};
  // layer 1: [64x32] @ [32x16]
  v16h bop = load_b_op(actA, 32, 0, n, hi);
  v8f h1[4];
#pragma unroll
  for (int t = 0; t < 4; ++t) h1[t] = wmma16(A1[t], bop, zero);
#pragma unroll
  for (int t = 0; t < 4; ++t) {
    v8f bb = load_bias_tile(bias, 0, t, hi);
    v8f v;
#pragma unroll
    for (int r = 0; r < 8; ++r) v[r] = softplus_f(h1[t][r] + bb[r]);
    store_act_tile(actB, 64, t, n, hi, v);
  }
  // layer 2: [64x64] @ [64x16], two K-chunks chained through C
  v16h c0 = load_b_op(actB, 64, 0, n, hi);
  v16h c1 = load_b_op(actB, 64, 1, n, hi);
  v8f h2[4];
#pragma unroll
  for (int t = 0; t < 4; ++t) {
    v8f acc = wmma16(A2[t][0], c0, zero);
    h2[t] = wmma16(A2[t][1], c1, acc);
  }
#pragma unroll
  for (int t = 0; t < 4; ++t) {
    v8f bb = load_bias_tile(bias, 64, t, hi);
    v8f v;
#pragma unroll
    for (int r = 0; r < 8; ++r) v[r] = softplus_f(h2[t][r] + bb[r]);
    store_act_tile(actB, 64, t, n, hi, v);  // in-order LDS: reads above already done
  }
  // layer 3: [32x64] @ [64x16]
  v16h d0 = load_b_op(actB, 64, 0, n, hi);
  v16h d1 = load_b_op(actB, 64, 1, n, hi);
#pragma unroll
  for (int t = 0; t < 2; ++t) {
    v8f acc = wmma16(A3[t][0], d0, zero);
    acc = wmma16(A3[t][1], d1, acc);
    v8f bb = load_bias_tile(bias, 128, t, hi);
#pragma unroll
    for (int r = 0; r < 8; ++r) acc[r] += bb[r];
    kstore(kbuf, s, t, n, hi, acc);
  }
}

// y_stage = y + hh * sum_j coef[j] * k_j ; written into actA as f16 B-layout
__device__ __forceinline__ void write_stage(_Float16* actA, const v8f y[2],
                                            const float* kbuf, float hh,
                                            const float* coef, int nk, int n,
                                            int hi) {
#pragma unroll
  for (int t = 0; t < 2; ++t) {
    v8f acc = y[t];
#pragma unroll 8
    for (int j = 0; j < nk; ++j) {
      v8f kv = kload(kbuf, j, t, n, hi);
      float c = hh * coef[j];
#pragma unroll
      for (int r = 0; r < 8; ++r) acc[r] = fmaf(c, kv[r], acc[r]);
    }
    store_act_tile(actA, 32, t, n, hi, acc);
  }
}

// ---------------- persistent kernel: full Dopri5 integration ----------------
__global__ __launch_bounds__(32) void node_dopri5_wmma(
    const float* __restrict__ ts, const float* __restrict__ y0,
    const float* __restrict__ W1, const float* __restrict__ b1,
    const float* __restrict__ W2, const float* __restrict__ b2,
    const float* __restrict__ W3, const float* __restrict__ b3,
    float* __restrict__ out) {
  __shared__ __align__(128) char smem[SMEM_BYTES];
  _Float16* actA = (_Float16*)(smem + ACTA_OFF);
  _Float16* actB = (_Float16*)(smem + ACTB_OFF);
  float* bias = (float*)(smem + BIAS_OFF);
  float* kbuf = (float*)(smem + KBUF_OFF);

  const int lane = threadIdx.x;
  const int n = lane & 15;    // batch column within tile / weight row within tile
  const int hi = lane >> 4;   // lane-half selector
  const int bt = blockIdx.x;  // batch tile: rows bt*16 .. bt*16+15

#if HAVE_TDM
  // Async-DMA the three bias vectors into LDS via the Tensor Data Mover.
  tdm_load_1d(b1, BIAS_OFF, 64);
  tdm_load_1d(b2, BIAS_OFF + 256, 64);
  tdm_load_1d(b3, BIAS_OFF + 512, 32);
#else
  for (int i = lane; i < 160; i += 32)
    bias[i] = (i < 64) ? b1[i] : (i < 128 ? b2[i - 64] : b3[i - 128]);
#endif

  // Weights -> registers as f16 A-operands (loop invariant, 128 VGPRs).
  v16h A1[4], A2[4][2], A3[2][2];
#pragma unroll
  for (int t = 0; t < 4; ++t) A1[t] = load_wtile(W1, 32, t, 0, n, hi);
#pragma unroll
  for (int t = 0; t < 4; ++t) {
    A2[t][0] = load_wtile(W2, 64, t, 0, n, hi);
    A2[t][1] = load_wtile(W2, 64, t, 1, n, hi);
  }
#pragma unroll
  for (int t = 0; t < 2; ++t) {
    A3[t][0] = load_wtile(W3, 64, t, 0, n, hi);
    A3[t][1] = load_wtile(W3, 64, t, 1, n, hi);
  }

  // y0 -> registers (C/D layout), and write ys[0].
  v8f y[2];
#pragma unroll
  for (int t = 0; t < 2; ++t) {
    const float* p = y0 + (size_t)(bt * 16 + n) * DD + t * 16 + hi * 8;
    v4f a = *(const v4f*)p;
    v4f b = *(const v4f*)(p + 4);
#pragma unroll
    for (int j = 0; j < 4; ++j) { y[t][j] = a[j]; y[t][4 + j] = b[j]; }
    float* op = out + (size_t)(bt * 16 + n) * DD + t * 16 + hi * 8;
    *(v4f*)op = a;
    *(v4f*)(op + 4) = b;
  }

#if HAVE_TDM
#if __has_builtin(__builtin_amdgcn_s_wait_tensorcnt)
  __builtin_amdgcn_s_wait_tensorcnt(0);
#endif
#endif

  // Dormand-Prince tableau
  const float a21 = 0.2f;
  const float a3[2] = {3.f / 40.f, 9.f / 40.f};
  const float a4[3] = {44.f / 45.f, -56.f / 15.f, 32.f / 9.f};
  const float a5[4] = {19372.f / 6561.f, -25360.f / 2187.f, 64448.f / 6561.f,
                       -212.f / 729.f};
  const float a6[5] = {9017.f / 3168.f, -355.f / 33.f, 46732.f / 5247.f,
                       49.f / 176.f, -5103.f / 18656.f};
  const int fs[5] = {0, 2, 3, 4, 5};
  const float fc[5] = {35.f / 384.f, 500.f / 1113.f, 125.f / 192.f,
                       -2187.f / 6784.f, 11.f / 84.f};

#pragma unroll 1
  for (int ii = 0; ii < TT - 1; ++ii) {
    float hh = (ts[ii + 1] - ts[ii]) * (1.0f / (float)KSUB);
#pragma unroll 1
    for (int sub = 0; sub < KSUB; ++sub) {
      write_stage(actA, y, kbuf, hh, (const float*)0, 0, n, hi);  // stage input = y
      feval(A1, A2, A3, actA, actB, bias, kbuf, 0, n, hi);        // k1
      {
        const float c[1] = {a21};
        write_stage(actA, y, kbuf, hh, c, 1, n, hi);
      }
      feval(A1, A2, A3, actA, actB, bias, kbuf, 1, n, hi);        // k2
      write_stage(actA, y, kbuf, hh, a3, 2, n, hi);
      feval(A1, A2, A3, actA, actB, bias, kbuf, 2, n, hi);        // k3
      write_stage(actA, y, kbuf, hh, a4, 3, n, hi);
      feval(A1, A2, A3, actA, actB, bias, kbuf, 3, n, hi);        // k4
      write_stage(actA, y, kbuf, hh, a5, 4, n, hi);
      feval(A1, A2, A3, actA, actB, bias, kbuf, 4, n, hi);        // k5
      write_stage(actA, y, kbuf, hh, a6, 5, n, hi);
      feval(A1, A2, A3, actA, actB, bias, kbuf, 5, n, hi);        // k6
      // 5th-order update: y += hh*(b1 k1 + b3 k3 + b4 k4 + b5 k5 + b6 k6)
#pragma unroll
      for (int t = 0; t < 2; ++t) {
        v8f acc = y[t];
#pragma unroll
        for (int j = 0; j < 5; ++j) {
          v8f kv = kload(kbuf, fs[j], t, n, hi);
          float c = hh * fc[j];
#pragma unroll
          for (int r = 0; r < 8; ++r) acc[r] = fmaf(c, kv[r], acc[r]);
        }
        y[t] = acc;
      }
    }
    // write ys[ii+1]
#pragma unroll
    for (int t = 0; t < 2; ++t) {
      float* op = out + ((size_t)(ii + 1) * BB + bt * 16 + n) * DD + t * 16 + hi * 8;
      v4f a, b;
#pragma unroll
      for (int j = 0; j < 4; ++j) { a[j] = y[t][j]; b[j] = y[t][4 + j]; }
      *(v4f*)op = a;
      *(v4f*)(op + 4) = b;
    }
  }

  // num_steps (int32) appended after ys, written as raw bits
  if (bt == 0 && lane == 0)
    ((int*)out)[(size_t)TT * BB * DD] = (TT - 1) * KSUB;
}

extern "C" void kernel_launch(void* const* d_in, const int* in_sizes, int n_in,
                              void* d_out, int out_size, void* d_ws,
                              size_t ws_size, hipStream_t stream) {
  const float* ts = (const float*)d_in[0];
  const float* y0 = (const float*)d_in[1];
  const float* W1 = (const float*)d_in[2];
  const float* b1 = (const float*)d_in[3];
  const float* W2 = (const float*)d_in[4];
  const float* b2 = (const float*)d_in[5];
  const float* W3 = (const float*)d_in[6];
  const float* b3 = (const float*)d_in[7];
  float* out = (float*)d_out;
  dim3 grid(BB / 16), block(32);
  hipLaunchKernelGGL(node_dopri5_wmma, grid, block, 0, stream, ts, y0, W1, b1,
                     W2, b2, W3, b3, out);
}
